// EncoderLayer_5325759447468
// MI455X (gfx1250) — compile-verified, matrix operand order checked by reading, not run on hardware
//
#include <hip/hip_runtime.h>
#include <hip/hip_bf16.h>

typedef __bf16 bf16;
typedef __attribute__((ext_vector_type(16))) __bf16 v16bf;
typedef __attribute__((ext_vector_type(8)))  __bf16 v8bf;
typedef __attribute__((ext_vector_type(8)))  float   v8f;

#define TOKS 4096   // B*S
#define DM   1024
#define DFF  4096
#define SEQ  2048
#define NH   16

__device__ __forceinline__ bf16 f2bf(float f) {
  unsigned u = __builtin_bit_cast(unsigned, f);
  u += 0x7FFFu + ((u >> 16) & 1u);           // round-to-nearest-even
  unsigned short h = (unsigned short)(u >> 16);
  return __builtin_bit_cast(bf16, h);
}

union FragU { v16bf v; v8bf h[2]; };

// Load a 16x32 operand fragment, CDNA5 16-bit layout:
// elements i=0..7 -> k = sub*8 + i ; i=8..15 -> k = sub*8 + 8 + i  (two b128s).
__device__ __forceinline__ v16bf ld_frag(const bf16* rowPtr, int sub) {
  FragU u;
  u.h[0] = *(const v8bf*)(rowPtr + sub * 8);
  u.h[1] = *(const v8bf*)(rowPtr + sub * 8 + 16);
  return u.v;
}

// ---------------------------------------------------------------------------
// fp32 -> bf16 elementwise convert
// ---------------------------------------------------------------------------
__global__ __launch_bounds__(256)
void cvt_bf16_kernel(const float* __restrict__ X, bf16* __restrict__ Y) {
  int i = (blockIdx.x * 256 + threadIdx.x) * 4;
  #pragma unroll
  for (int j = 0; j < 4; ++j) Y[i + j] = f2bf(X[i + j]);
}

// ---------------------------------------------------------------------------
// fp32 [K x N] -> bf16 transposed [N x K]
// ---------------------------------------------------------------------------
__global__ __launch_bounds__(256)
void transpose_cvt_kernel(const float* __restrict__ W, bf16* __restrict__ Wt,
                          int K, int N) {
  __shared__ float t[32][33];
  int bn = blockIdx.x * 32, bk = blockIdx.y * 32;
  int x = threadIdx.x, y = threadIdx.y;   // block (32,8)
  #pragma unroll
  for (int i = 0; i < 32; i += 8)
    t[y + i][x] = W[(size_t)(bk + y + i) * N + bn + x];
  __syncthreads();
  #pragma unroll
  for (int i = 0; i < 32; i += 8)
    Wt[(size_t)(bn + y + i) * K + bk + x] = f2bf(t[x][y + i]);
}

// ---------------------------------------------------------------------------
// bf16 GEMM, LDS double-buffered: C[M,N] = A[M,K] @ Bt[N,K]^T + bias.
// MODE 0: fp32 row-major.  MODE 1: bf16 scatter to [b,h,s,d].
// MODE 2: bf16 scatter to [b,h,d,s] (V transposed).  MODE 3: bf16 row-major.
// ---------------------------------------------------------------------------
template<int MODE, bool RELU>
__global__ __launch_bounds__(256)
void gemm_bf16(const bf16* __restrict__ A, const bf16* __restrict__ Bt,
               const float* __restrict__ bias,
               float* __restrict__ outF, bf16* __restrict__ outB,
               int M, int N, int K)
{
  constexpr int LD = 48;                   // 96B row stride: 16B aligned
  __shared__ bf16 As[2][128 * LD];
  __shared__ bf16 Bs[2][128 * LD];
  const int tid  = threadIdx.x;
  const int lane = tid & 31, w = tid >> 5;
  const int wm = w & 3, wn = w >> 2;       // 4 waves along M, 2 along N
  const int sub = lane >> 4, l16 = lane & 15;
  const int m0 = blockIdx.y * 128, n0 = blockIdx.x * 128;

  // this thread's two 16B chunks of the 128x32 tile (512 chunks total)
  const int r0 = tid >> 2,         kc0 = (tid & 3) * 8;
  const int r1 = (tid + 256) >> 2, kc1 = ((tid + 256) & 3) * 8;

  const v8f vz = {0.f,0.f,0.f,0.f,0.f,0.f,0.f,0.f};
  v8f acc[2][4];
  #pragma unroll
  for (int i = 0; i < 2; ++i)
    #pragma unroll
    for (int j = 0; j < 4; ++j) acc[i][j] = vz;

  v8bf ra[2], rb[2];
  auto gload = [&](int k0) {
    ra[0] = *(const v8bf*)(A  + (size_t)(m0 + r0) * K + k0 + kc0);
    rb[0] = *(const v8bf*)(Bt + (size_t)(n0 + r0) * K + k0 + kc0);
    ra[1] = *(const v8bf*)(A  + (size_t)(m0 + r1) * K + k0 + kc1);
    rb[1] = *(const v8bf*)(Bt + (size_t)(n0 + r1) * K + k0 + kc1);
    __builtin_prefetch(A  + (size_t)(m0 + r0) * K + k0 + 32 + kc0, 0, 1);
    __builtin_prefetch(Bt + (size_t)(n0 + r0) * K + k0 + 32 + kc0, 0, 1);
  };
  auto sstore = [&](int b) {
    *(v8bf*)(As[b] + r0 * LD + kc0) = ra[0];
    *(v8bf*)(Bs[b] + r0 * LD + kc0) = rb[0];
    *(v8bf*)(As[b] + r1 * LD + kc1) = ra[1];
    *(v8bf*)(Bs[b] + r1 * LD + kc1) = rb[1];
  };

  gload(0);
  sstore(0);
  __syncthreads();

  const int T = K >> 5;
  for (int t = 0; t < T; ++t) {
    const int cur = t & 1, nxt = cur ^ 1;
    if (t + 1 < T) gload((t + 1) << 5);    // overlap global latency with WMMA
    v16bf af[2], bfg[4];
    #pragma unroll
    for (int fa = 0; fa < 2; ++fa)
      af[fa] = ld_frag(As[cur] + (wm * 32 + fa * 16 + l16) * LD, sub);
    #pragma unroll
    for (int fn = 0; fn < 4; ++fn)
      bfg[fn] = ld_frag(Bs[cur] + (wn * 64 + fn * 16 + l16) * LD, sub);
    #pragma unroll
    for (int fa = 0; fa < 2; ++fa)
      #pragma unroll
      for (int fn = 0; fn < 4; ++fn)
        acc[fa][fn] = __builtin_amdgcn_wmma_f32_16x16x32_bf16(
            false, af[fa], false, bfg[fn], (short)0, acc[fa][fn], false, false);
    if (t + 1 < T) sstore(nxt);
    __syncthreads();
  }

  #pragma unroll
  for (int fa = 0; fa < 2; ++fa) {
    #pragma unroll
    for (int fn = 0; fn < 4; ++fn) {
      const int n  = n0 + wn * 64 + fn * 16 + l16;
      const float bv = bias[n];
      #pragma unroll
      for (int j = 0; j < 8; ++j) {
        const int m = m0 + wm * 32 + fa * 16 + sub * 8 + j;
        float v = acc[fa][fn][j] + bv;
        if (RELU) v = fmaxf(v, 0.f);
        if (MODE == 0) {
          outF[(size_t)m * N + n] = v;
        } else if (MODE == 1) {
          int b = m >> 11, s = m & 2047, h = n >> 6, d = n & 63;
          outB[((((size_t)b * NH + h) * SEQ + s) << 6) + d] = f2bf(v);
        } else if (MODE == 2) {
          int b = m >> 11, s = m & 2047, h = n >> 6, d = n & 63;
          outB[(((size_t)b * NH + h) * 64 + d) * SEQ + s] = f2bf(v);
        } else {
          outB[(size_t)m * N + n] = f2bf(v);
        }
      }
    }
  }
}

// ---------------------------------------------------------------------------
// Flash attention: Q,K bf16 [BH][S][64], VT bf16 [BH][64][S] ->
// O bf16 [B][S][D_MODEL] (head-interleaved row-major).
// Block = 8 waves x 16 query rows; K/V^T blocks staged cooperatively in LDS
// (8x traffic reduction); online softmax; WMMA for QK^T and P*V.
// ---------------------------------------------------------------------------
__global__ __launch_bounds__(256)
void attention_kernel(const bf16* __restrict__ Q, const bf16* __restrict__ Kb,
                      const bf16* __restrict__ VT, bf16* __restrict__ O)
{
  constexpr int HD = 64, KLD = 80, PLD = 80;  // 160B strides: 16B aligned
  __shared__ bf16 Ks[64 * KLD];               // 64 keys x 64 d
  __shared__ bf16 Vs[64 * KLD];               // 64 d    x 64 keys (V^T)
  __shared__ bf16 Pst[8 * 16 * PLD];          // per-wave P staging
  const int tid  = threadIdx.x;
  const int lane = tid & 31, w = tid >> 5;
  const int sub = lane >> 4, l16 = lane & 15;
  const int bh = blockIdx.y;
  const int q0 = blockIdx.x * 128 + w * 16;
  const bf16* q  = Q  + (size_t)bh * SEQ * HD;
  const bf16* kp = Kb + (size_t)bh * SEQ * HD;
  const bf16* vt = VT + (size_t)bh * HD * SEQ;
  bf16* pst = Pst + w * 16 * PLD;
  const float scale = 0.125f;                 // 1/sqrt(64)

  // resident Q fragments (16 rows x 64 d = 2 A-frags)
  v16bf qf[2];
  #pragma unroll
  for (int fa = 0; fa < 2; ++fa)
    qf[fa] = ld_frag(q + (size_t)(q0 + l16) * HD + fa * 32, sub);

  const v8f vz = {0.f,0.f,0.f,0.f,0.f,0.f,0.f,0.f};
  v8f o[4];
  #pragma unroll
  for (int f = 0; f < 4; ++f) o[f] = vz;
  float rowM[8], rowL[8];
  #pragma unroll
  for (int j = 0; j < 8; ++j) { rowM[j] = -1e30f; rowL[j] = 0.f; }

  for (int kb = 0; kb < SEQ; kb += 64) {
    __syncthreads();                          // WAR on Ks/Vs/Pst
    // ---- cooperative K / V^T block staging (coalesced 16B chunks) ----
    #pragma unroll
    for (int i = 0; i < 2; ++i) {
      int c = tid + i * 256, r = c >> 3, dc = (c & 7) * 8;   // 512 chunks
      *(v8bf*)(Ks + r * KLD + dc) = *(const v8bf*)(kp + (size_t)(kb + r) * HD + dc);
      *(v8bf*)(Vs + r * KLD + dc) = *(const v8bf*)(vt + (size_t)r * SEQ + kb + dc);
    }
    __syncthreads();
    // ---- scores: 4 n-frags x (2 k-steps over d), K from LDS ----
    v8f sf[4];
    #pragma unroll
    for (int fn = 0; fn < 4; ++fn) {
      v8f a = vz;
      #pragma unroll
      for (int fa = 0; fa < 2; ++fa) {
        v16bf kf = ld_frag(Ks + (fn * 16 + l16) * KLD + fa * 32, sub);
        a = __builtin_amdgcn_wmma_f32_16x16x32_bf16(
                false, qf[fa], false, kf, (short)0, a, false, false);
      }
      sf[fn] = a;
    }
    // ---- online softmax (per row j; lanes within a half share rows) ----
    #pragma unroll
    for (int j = 0; j < 8; ++j) {
      float mx = fmaxf(fmaxf(sf[0][j], sf[1][j]), fmaxf(sf[2][j], sf[3][j])) * scale;
      #pragma unroll
      for (int off = 8; off; off >>= 1) mx = fmaxf(mx, __shfl_xor(mx, off, 16));
      float mnew  = fmaxf(rowM[j], mx);
      float alpha = __expf(rowM[j] - mnew);
      rowM[j] = mnew;
      rowL[j] *= alpha;
      #pragma unroll
      for (int f = 0; f < 4; ++f) o[f][j] *= alpha;
      float ps = 0.f;
      #pragma unroll
      for (int f = 0; f < 4; ++f) {
        float p = __expf(sf[f][j] * scale - mnew);
        ps += p;
        pst[(sub * 8 + j) * PLD + f * 16 + l16] = f2bf(p);   // C-layout -> LDS
      }
      rowL[j] += ps;
    }
    __syncthreads();                          // P visible (cross-lane exchange)
    // ---- P (A-layout) @ V, V^T from LDS ----
    v16bf pa[2];
    #pragma unroll
    for (int fk = 0; fk < 2; ++fk)
      pa[fk] = ld_frag(pst + l16 * PLD + fk * 32, sub);
    #pragma unroll
    for (int f = 0; f < 4; ++f) {
      #pragma unroll
      for (int fk = 0; fk < 2; ++fk) {
        v16bf vf = ld_frag(Vs + (f * 16 + l16) * KLD + fk * 32, sub);
        o[f] = __builtin_amdgcn_wmma_f32_16x16x32_bf16(
                   false, pa[fk], false, vf, (short)0, o[f], false, false);
      }
    }
  }

  // ---- finalize: divide by l, store head-interleaved bf16 ----
  const int b = bh >> 4, h = bh & 15;
  #pragma unroll
  for (int j = 0; j < 8; ++j) {
    float l = rowL[j];
    #pragma unroll
    for (int off = 8; off; off >>= 1) l += __shfl_xor(l, off, 16);
    float inv = 1.0f / l;
    size_t row = ((size_t)b * SEQ + q0 + sub * 8 + j) * DM + h * 64;
    #pragma unroll
    for (int f = 0; f < 4; ++f)
      O[row + f * 16 + l16] = f2bf(o[f][j] * inv);
  }
}

// ---------------------------------------------------------------------------
// y = LayerNorm(X + R) * g + b ; fp32 out (+ optional bf16 out)
// ---------------------------------------------------------------------------
__global__ __launch_bounds__(256)
void ln_kernel(const float* __restrict__ X, const float* __restrict__ R,
               const float* __restrict__ g, const float* __restrict__ bb,
               float* __restrict__ outF, bf16* __restrict__ outB)
{
  constexpr int D = DM;
  const int row = blockIdx.x, tid = threadIdx.x;
  __shared__ float red[8];
  const float* x = X + (size_t)row * D;
  const float* r = R + (size_t)row * D;
  float v[4]; float s = 0.f;
  #pragma unroll
  for (int i = 0; i < 4; ++i) { int c = tid + i * 256; v[i] = x[c] + r[c]; s += v[i]; }
  #pragma unroll
  for (int off = 16; off; off >>= 1) s += __shfl_xor(s, off, 32);
  if ((tid & 31) == 0) red[tid >> 5] = s;
  __syncthreads();
  float mu = 0.f;
  #pragma unroll
  for (int i = 0; i < 8; ++i) mu += red[i];
  mu *= (1.0f / D);
  __syncthreads();
  float vs = 0.f;
  #pragma unroll
  for (int i = 0; i < 4; ++i) { float d = v[i] - mu; vs += d * d; }
  #pragma unroll
  for (int off = 16; off; off >>= 1) vs += __shfl_xor(vs, off, 32);
  if ((tid & 31) == 0) red[tid >> 5] = vs;
  __syncthreads();
  float var = 0.f;
  #pragma unroll
  for (int i = 0; i < 8; ++i) var += red[i];
  var *= (1.0f / D);
  float rs = rsqrtf(var + 1e-5f);
  #pragma unroll
  for (int i = 0; i < 4; ++i) {
    int c = tid + i * 256;
    float y = (v[i] - mu) * rs * g[c] + bb[c];
    outF[(size_t)row * D + c] = y;
    if (outB) outB[(size_t)row * D + c] = f2bf(y);
  }
}

// ---------------------------------------------------------------------------
extern "C" void kernel_launch(void* const* d_in, const int* in_sizes, int n_in,
                              void* d_out, int out_size, void* d_ws, size_t ws_size,
                              hipStream_t stream)
{
  const float* x    = (const float*)d_in[0];
  const float* Wq   = (const float*)d_in[1];
  const float* bq   = (const float*)d_in[2];
  const float* Wk   = (const float*)d_in[3];
  const float* bk   = (const float*)d_in[4];
  const float* Wv   = (const float*)d_in[5];
  const float* bv   = (const float*)d_in[6];
  const float* Wo   = (const float*)d_in[7];
  const float* bo   = (const float*)d_in[8];
  const float* W1   = (const float*)d_in[9];
  const float* b1   = (const float*)d_in[10];
  const float* W2   = (const float*)d_in[11];
  const float* b2   = (const float*)d_in[12];
  const float* ln1g = (const float*)d_in[13];
  const float* ln1b = (const float*)d_in[14];
  const float* ln2g = (const float*)d_in[15];
  const float* ln2b = (const float*)d_in[16];

  char* p = (char*)d_ws;
  auto carve = [&](size_t bytes) { void* r = (void*)p; p += (bytes + 255) & ~(size_t)255; return r; };
  bf16*  xb   = (bf16*) carve((size_t)TOKS * DM * 2);
  bf16*  Wqt  = (bf16*) carve((size_t)DM * DM * 2);
  bf16*  Wkt  = (bf16*) carve((size_t)DM * DM * 2);
  bf16*  Wvt  = (bf16*) carve((size_t)DM * DM * 2);
  bf16*  Wot  = (bf16*) carve((size_t)DM * DM * 2);
  bf16*  W1t  = (bf16*) carve((size_t)DM * DFF * 2);
  bf16*  W2t  = (bf16*) carve((size_t)DFF * DM * 2);
  bf16*  qh   = (bf16*) carve((size_t)TOKS * DM * 2);   // [b,h,s,d]
  bf16*  kh   = (bf16*) carve((size_t)TOKS * DM * 2);   // [b,h,s,d]
  bf16*  vth  = (bf16*) carve((size_t)TOKS * DM * 2);   // [b,h,d,s]
  bf16*  attnO= (bf16*) carve((size_t)TOKS * DM * 2);   // [b,s,D]
  float* aprj = (float*)carve((size_t)TOKS * DM * 4);
  float* x1f  = (float*)carve((size_t)TOKS * DM * 4);
  bf16*  x1b  = (bf16*) carve((size_t)TOKS * DM * 2);
  bf16*  hb   = qh;     // 32MB alias over qh/kh/vth/attnO (dead after Wo gemm)
  float* ff   = aprj;   // alias: attn_proj dead after LN1

  // --- precision conversion / weight transposes ---
  cvt_bf16_kernel<<<TOKS * DM / 1024, 256, 0, stream>>>(x, xb);
  transpose_cvt_kernel<<<dim3(DM / 32, DM / 32),  dim3(32, 8), 0, stream>>>(Wq, Wqt, DM, DM);
  transpose_cvt_kernel<<<dim3(DM / 32, DM / 32),  dim3(32, 8), 0, stream>>>(Wk, Wkt, DM, DM);
  transpose_cvt_kernel<<<dim3(DM / 32, DM / 32),  dim3(32, 8), 0, stream>>>(Wv, Wvt, DM, DM);
  transpose_cvt_kernel<<<dim3(DM / 32, DM / 32),  dim3(32, 8), 0, stream>>>(Wo, Wot, DM, DM);
  transpose_cvt_kernel<<<dim3(DFF / 32, DM / 32), dim3(32, 8), 0, stream>>>(W1, W1t, DM, DFF);
  transpose_cvt_kernel<<<dim3(DM / 32, DFF / 32), dim3(32, 8), 0, stream>>>(W2, W2t, DFF, DM);

  // --- QKV projections (bf16 WMMA, head-layout scatter epilogues) ---
  gemm_bf16<1, false><<<dim3(DM / 128, TOKS / 128), 256, 0, stream>>>(
      xb, Wqt, bq, nullptr, qh, TOKS, DM, DM);
  gemm_bf16<1, false><<<dim3(DM / 128, TOKS / 128), 256, 0, stream>>>(
      xb, Wkt, bk, nullptr, kh, TOKS, DM, DM);
  gemm_bf16<2, false><<<dim3(DM / 128, TOKS / 128), 256, 0, stream>>>(
      xb, Wvt, bv, nullptr, vth, TOKS, DM, DM);

  // --- fused flash attention ---
  attention_kernel<<<dim3(SEQ / 128, 32), 256, 0, stream>>>(qh, kh, vth, attnO);

  // --- output projection + LN1 ---
  gemm_bf16<0, false><<<dim3(DM / 128, TOKS / 128), 256, 0, stream>>>(
      attnO, Wot, bo, aprj, nullptr, TOKS, DM, DM);
  ln_kernel<<<TOKS, 256, 0, stream>>>(x, aprj, ln1g, ln1b, x1f, x1b);

  // --- FFN ---
  gemm_bf16<3, true><<<dim3(DFF / 128, TOKS / 128), 256, 0, stream>>>(
      x1b, W1t, b1, nullptr, hb, TOKS, DFF, DM);
  gemm_bf16<0, false><<<dim3(DM / 128, TOKS / 128), 256, 0, stream>>>(
      hb, W2t, b2, ff, nullptr, TOKS, DM, DFF);

  // --- LN2 -> final fp32 output ---
  ln_kernel<<<TOKS, 256, 0, stream>>>(x1f, ff, ln2g, ln2b, (float*)d_out, nullptr);

  (void)in_sizes; (void)n_in; (void)out_size; (void)ws_size;
}
